// CoCa_images_55422257987716
// MI455X (gfx1250) — matrix operand order repaired and checked by reading, not compiled, not run on hardware
//
#include <hip/hip_runtime.h>
#include <hip/hip_bf16.h>

// ---------------- model constants ----------------
constexpr int kD     = 386;
constexpr int kH     = 8;
constexpr int kDH    = 64;
constexpr int kFF    = 1544;      // 4*D
constexpr int kFUSED = kH*kDH + kDH + kDH + 2*kFF;   // 3728
constexpr int kS     = 200;
constexpr int kN     = 201;       // seq len = S + cls
constexpr int kB     = 4;
constexpr int kM     = kB * kN;   // 804 rows of activations

// ---------------- WMMA vector types ----------------
typedef __attribute__((ext_vector_type(16))) _Float16 v16h;
typedef __attribute__((ext_vector_type(8)))  _Float16 v8h;
typedef __attribute__((ext_vector_type(8)))  float    v8f;

// ---------------- CDNA5 async global->LDS path (guarded) ----------------
#if __has_builtin(__builtin_amdgcn_global_load_async_to_lds_b64) && \
    __has_builtin(__builtin_amdgcn_s_wait_asynccnt)
#define USE_ASYNC_LDS 1
// builtin signature: void(v2i addrspace(1)*, v2i addrspace(3)*, imm offset, imm cpol)
typedef int v2i_t __attribute__((ext_vector_type(2)));
typedef __attribute__((address_space(1))) v2i_t gas_v2i;
typedef __attribute__((address_space(3))) v2i_t las_v2i;
#else
#define USE_ASYNC_LDS 0
#endif

// =====================================================================
// Generic WMMA GEMM: C[M,N] = A[M,K] (fp32) @ B[K,N] (fp32), f16 WMMA,
// fp32 accumulate. mode: 0 = C=AB ; 1 = C=AB+R ; 2 = C+=AB (reads C).
// Block tile 128x128, K-step 32, 8 waves (256 thr); wave w owns rows
// [16w,16w+16) x all 128 cols (8 accumulators of 16x16, A frag reused).
// B (weight) tiles staged memory->LDS via GLOBAL_LOAD_ASYNC_TO_LDS_B64
// when available, then converted fp32->f16 into WMMA fragment order.
// =====================================================================
#define BM 128
#define BN 128
#define BK 32

__global__ __launch_bounds__(256)
void gemm_wmma_f16(const float* __restrict__ A, const float* __restrict__ B,
                   float* __restrict__ C, const float* __restrict__ R,
                   int M, int N, int K, int mode) {
  __shared__ _Float16 As[BM * BK];   // [m][k], pitch 32 halves (64B rows)
  __shared__ _Float16 BsT[BN * BK];  // transposed [n][k], pitch 32 halves
#if USE_ASYNC_LDS
  __shared__ float Bstage[BK * BN];  // raw fp32 [k][n] tile, async-filled
#endif

  const int tid  = threadIdx.x;
  const int wave = tid >> 5;
  const int lane = tid & 31;
  const int m0 = blockIdx.y * BM;
  const int n0 = blockIdx.x * BN;

  v8f acc[8] = {};

  const int lr = lane & 15;      // row/col within 16
  const int hl = lane >> 4;      // half-wave select

  for (int k0 = 0; k0 < K; k0 += BK) {
    // --- prefetch next A K-tile into GL2 (global_prefetch_b8) ---
    if (k0 + BK < K) {
      __builtin_prefetch(&A[(size_t)(m0 + (tid >> 1)) * K + k0 + BK], 0, 0);
    }
    // --- stage A tile (128x32), coalesced over K, fp32->f16 in VGPRs ---
    #pragma unroll
    for (int e = tid; e < BM * BK; e += 256) {
      int r = e >> 5, c = e & 31;
      int gm = m0 + r, gk = k0 + c;
      float v = (gm < M && gk < K) ? A[(size_t)gm * K + gk] : 0.0f;
      As[e] = (_Float16)v;
    }

#if USE_ASYNC_LDS
    // --- B tile: async DMA memory -> LDS staging (b64 = 2 fp32/lane-op).
    //     All N in this model are even and rows are 8B aligned. ---
    #pragma unroll
    for (int e = tid; e < BK * (BN / 2); e += 256) {
      int kk = e >> 6, cc = e & 63;          // 64 chunks of 8B per tile row
      int gk = k0 + kk, gn = n0 + cc * 2;
      if (gk < K && gn < N) {
        const float* gp = B + (size_t)gk * N + gn;
        __builtin_amdgcn_global_load_async_to_lds_b64(
            (gas_v2i*)gp, (las_v2i*)&Bstage[kk * BN + cc * 2],
            /*offset=*/0, /*cpol=*/0);
      }
    }
    __builtin_amdgcn_s_wait_asynccnt(0);
    __syncthreads();                          // async data + As visible
    // --- convert/transpose staging fp32 [k][n] -> BsT f16 [n][k] ---
    #pragma unroll
    for (int e = tid; e < BN * BK; e += 256) {
      int kk = e >> 7, nn = e & 127;          // coalesced read over n
      int gk = k0 + kk, gn = n0 + nn;
      float v = (gk < K && gn < N) ? Bstage[kk * BN + nn] : 0.0f;
      BsT[nn * BK + kk] = (_Float16)v;
    }
#else
    // --- fallback: stage B tile transposed via VGPRs ---
    #pragma unroll
    for (int e = tid; e < BN * BK; e += 256) {
      int kk = e >> 7, nn = e & 127;
      int gk = k0 + kk, gn = n0 + nn;
      float v = (gk < K && gn < N) ? B[(size_t)gk * N + gn] : 0.0f;
      BsT[nn * BK + kk] = (_Float16)v;
    }
#endif
    __syncthreads();

    // --- A fragment: lane holds row (16w + lr); halves 0..7 = K hl*8..+7,
    //     halves 8..15 = K hl*8+16..+23 (ISA 16-bit A 16x32 layout) ---
    const _Float16* ap = &As[(wave * 16 + lr) * BK + (hl << 3)];
    v8h alo = *reinterpret_cast<const v8h*>(ap);
    v8h ahi = *reinterpret_cast<const v8h*>(ap + 16);
    v16h afrag = __builtin_shufflevector(alo, ahi,
        0,1,2,3,4,5,6,7,8,9,10,11,12,13,14,15);

    #pragma unroll
    for (int nb = 0; nb < 8; ++nb) {
      // --- B fragment: lane holds col (nb*16 + lr); halves 0..7 -> K hl*8+j, 8..15 -> K hl*8+16+(j-8) ---
      const _Float16* bp = &BsT[(nb * 16 + lr) * BK + (hl << 3)];
      v8h blo = *reinterpret_cast<const v8h*>(bp);
      v8h bhi = *reinterpret_cast<const v8h*>(bp + 16);
      v16h bfrag = __builtin_shufflevector(blo, bhi,
          0,1,2,3,4,5,6,7,8,9,10,11,12,13,14,15);

      acc[nb] = __builtin_amdgcn_wmma_f32_16x16x32_f16(
          /*neg_a=*/false, afrag, /*neg_b=*/false, bfrag,
          /*c_mod=*/(short)0, acc[nb], /*reuse_a=*/false, /*reuse_b=*/false);
    }
    __syncthreads();
  }

  // --- epilogue: C/D layout: lanes 0-15 -> M=r, lanes 16-31 -> M=8+r ---
  const int mb = hl * 8;
  #pragma unroll
  for (int nb = 0; nb < 8; ++nb) {
    #pragma unroll
    for (int r = 0; r < 8; ++r) {
      int gm = m0 + wave * 16 + mb + r;
      int gn = n0 + nb * 16 + lr;
      if (gm < M && gn < N) {
        float v = acc[nb][r];
        if (mode == 1)      v += R[(size_t)gm * N + gn];
        else if (mode == 2) v += C[(size_t)gm * N + gn];
        C[(size_t)gm * N + gn] = v;
      }
    }
  }
}

// =====================================================================
// Patch embed: conv 3x3 stride 2 VALID (3->386) + bias + relu + 19x19 mean.
// One workgroup per image, image staged in LDS. half selects first/second
// S images of each batch element. Also writes the cls token row (s==0 blk).
// =====================================================================
__global__ __launch_bounds__(256)
void embed_kernel(const float* __restrict__ img, const float* __restrict__ w,
                  const float* __restrict__ bias, const float* __restrict__ cls,
                  float* __restrict__ out, int half) {
  __shared__ float simg[3 * 40 * 40];
  const int blk = blockIdx.x;
  const int bi = blk / kS;
  const int s  = blk % kS;
  const float* ip = img + ((size_t)bi * 2 * kS + (size_t)half * kS + s) * 3 * 40 * 40;
  for (int e = threadIdx.x; e < 4800; e += 256) simg[e] = ip[e];
  __syncthreads();

  for (int c = threadIdx.x; c < kD; c += 256) {
    float wl[27];
    #pragma unroll
    for (int t = 0; t < 27; ++t) wl[t] = w[c * 27 + t];
    const float bb = bias[c];
    float total = 0.0f;
    for (int oy = 0; oy < 19; ++oy) {
      for (int ox = 0; ox < 19; ++ox) {
        float sacc = bb;
        #pragma unroll
        for (int ci = 0; ci < 3; ++ci)
          #pragma unroll
          for (int ky = 0; ky < 3; ++ky)
            #pragma unroll
            for (int kx = 0; kx < 3; ++kx)
              sacc += simg[ci * 1600 + (2 * oy + ky) * 40 + (2 * ox + kx)]
                    * wl[ci * 9 + ky * 3 + kx];
        total += fmaxf(sacc, 0.0f);
      }
    }
    out[((size_t)bi * kN + s) * kD + c] = total * (1.0f / 361.0f);
  }
  if (s == 0) {
    for (int c = threadIdx.x; c < kD; c += 256)
      out[((size_t)bi * kN + kS) * kD + c] = cls[c];
  }
}

// =====================================================================
// LayerNorm over D per row (x (rows,D) -> y), gain g.
// =====================================================================
__global__ __launch_bounds__(256)
void ln_kernel(const float* __restrict__ x, const float* __restrict__ g,
               float* __restrict__ y) {
  __shared__ float red[2][8];
  const size_t row = blockIdx.x;
  const float* xr = x + row * kD;
  float s = 0.f, s2 = 0.f;
  for (int c = threadIdx.x; c < kD; c += 256) { float v = xr[c]; s += v; s2 += v * v; }
  for (int off = 16; off; off >>= 1) { s += __shfl_xor(s, off, 32); s2 += __shfl_xor(s2, off, 32); }
  const int wv = threadIdx.x >> 5, lane = threadIdx.x & 31;
  if (lane == 0) { red[0][wv] = s; red[1][wv] = s2; }
  __syncthreads();
  if (threadIdx.x == 0) {
    float a = 0.f, b = 0.f;
    for (int i = 0; i < 8; ++i) { a += red[0][i]; b += red[1][i]; }
    red[0][0] = a; red[1][0] = b;
  }
  __syncthreads();
  s = red[0][0]; s2 = red[1][0];
  const float m = s / kD;
  const float var = s2 / kD - m * m;
  const float inv = rsqrtf(var + 1e-5f);
  for (int c = threadIdx.x; c < kD; c += 256)
    y[row * kD + c] = (xr[c] - m) * inv * g[c];
}

// =====================================================================
// Extract q/k/v from fused proj (row-major (M, FUSED)) and apply rotary.
// q: cols [0,512), k: [512,576), v: [576,640). One block of 64 per row.
// =====================================================================
__global__ __launch_bounds__(64)
void extract_qkv_kernel(const float* __restrict__ fused, float* __restrict__ qr,
                        float* __restrict__ kr, float* __restrict__ vb) {
  const size_t row = blockIdx.x;
  const int i = (int)(row % kN);
  const int d = threadIdx.x;
  const int t = d & 31;
  const float inv = __expf(-((float)(2 * t) / 64.0f) * 9.210340371976184f); // ln(1e4)
  const float fr = (float)i * inv;
  const float c = __cosf(fr), sn = __sinf(fr);
  const float sign = (d < 32) ? -1.0f : 1.0f;
  const float* f0 = fused + row * kFUSED;

  float kd = f0[512 + d], ko = f0[512 + (d ^ 32)];
  kr[row * 64 + d] = kd * c + sign * ko * sn;
  vb[row * 64 + d] = f0[576 + d];
  #pragma unroll
  for (int h = 0; h < kH; ++h) {
    float qd = f0[h * 64 + d], qo = f0[h * 64 + (d ^ 32)];
    qr[row * 512 + h * 64 + d] = qd * c + sign * qo * sn;
  }
}

// =====================================================================
// Multi-query attention, one wave per (b, h, i): online softmax over j.
// k/v have 1 head (pitch kp/vp). q (M,512). causal => j<=i.
// =====================================================================
__global__ __launch_bounds__(32)
void attn_kernel(const float* __restrict__ q, const float* __restrict__ k,
                 const float* __restrict__ v, float* __restrict__ out,
                 int causal, int kp, int vp) {
  const int blk = blockIdx.x;
  const int i  = blk % kN;
  const int h  = (blk / kN) % kH;
  const int bi = blk / (kN * kH);
  const int lane = threadIdx.x;
  const size_t rowq = ((size_t)bi * kN + i) * 512 + h * 64;
  const float qa = q[rowq + lane];
  const float qb = q[rowq + lane + 32];
  float mx = -3.0e38f, l = 0.f, acca = 0.f, accb = 0.f;
  const int jmax = causal ? i : (kN - 1);
  const size_t base = (size_t)bi * kN;
  for (int j = 0; j <= jmax; ++j) {
    const float* kj = k + (base + j) * kp;
    float p = qa * kj[lane] + qb * kj[lane + 32];
    for (int off = 16; off; off >>= 1) p += __shfl_xor(p, off, 32);
    const float sc = p * 0.125f;                    // DH^-0.5
    const float mn = fmaxf(mx, sc);
    const float corr = __expf(mx - mn);
    const float pe = __expf(sc - mn);
    const float* vj = v + (base + j) * vp;
    acca = acca * corr + pe * vj[lane];
    accb = accb * corr + pe * vj[lane + 32];
    l = l * corr + pe;
    mx = mn;
  }
  out[rowq + lane]      = acca / l;
  out[rowq + lane + 32] = accb / l;
}

// =====================================================================
// SwiGLU gate: dst[r,c] = x1 * silu(gate), both read from src with stride.
// =====================================================================
__global__ void silu_mul_kernel(const float* __restrict__ src, float* __restrict__ dst,
                                int rows, int F, int stride, int x1off, int goff) {
  const size_t idx = (size_t)blockIdx.x * blockDim.x + threadIdx.x;
  if (idx >= (size_t)rows * F) return;
  const size_t r = idx / F;
  const int c = (int)(idx % F);
  const float x1 = src[r * stride + x1off + c];
  const float g  = src[r * stride + goff + c];
  dst[r * F + c] = x1 * g / (1.0f + __expf(-g));
}

// =====================================================================
// Final: LN(cls) @ logit_w -> (4,2). One wave per batch row.
// =====================================================================
__global__ __launch_bounds__(128)
void logit_kernel(const float* __restrict__ a, const float* __restrict__ g,
                  const float* __restrict__ w, float* __restrict__ out) {
  const int wv = threadIdx.x >> 5;
  const int lane = threadIdx.x & 31;
  if (wv >= kB) return;
  const float* row = a + ((size_t)wv * kN + kS) * kD;   // cls = token S
  float s = 0.f, s2 = 0.f;
  for (int c = lane; c < kD; c += 32) { float v = row[c]; s += v; s2 += v * v; }
  for (int off = 16; off; off >>= 1) { s += __shfl_xor(s, off, 32); s2 += __shfl_xor(s2, off, 32); }
  const float m = s / kD;
  const float inv = rsqrtf(s2 / kD - m * m + 1e-5f);
  float d0 = 0.f, d1 = 0.f;
  for (int c = lane; c < kD; c += 32) {
    float xn = (row[c] - m) * inv * g[c];
    d0 += xn * w[c * 2 + 0];
    d1 += xn * w[c * 2 + 1];
  }
  for (int off = 16; off; off >>= 1) { d0 += __shfl_xor(d0, off, 32); d1 += __shfl_xor(d1, off, 32); }
  if (lane == 0) { out[wv * 2 + 0] = d0; out[wv * 2 + 1] = d1; }
}

// =====================================================================
// Host orchestration
// =====================================================================
extern "C" void kernel_launch(void* const* d_in, const int* in_sizes, int n_in,
                              void* d_out, int out_size, void* d_ws, size_t ws_size,
                              hipStream_t stream) {
  (void)in_sizes; (void)n_in; (void)out_size; (void)ws_size;

  const float* input     = (const float*)d_in[0];
  const float* conv_w    = (const float*)d_in[1];
  const float* conv_b    = (const float*)d_in[2];
  const float* cls_token = (const float*)d_in[3];
  const float* uni_ln    = (const float*)d_in[4];
  const float* uni_fused = (const float*)d_in[5];
  const float* uni_aout  = (const float*)d_in[6];
  const float* uni_fout  = (const float*)d_in[7];
  const float* mm_ln     = (const float*)d_in[8];
  const float* mm_fused  = (const float*)d_in[9];
  const float* mm_aout   = (const float*)d_in[10];
  const float* mm_fout   = (const float*)d_in[11];
  const float* ca_ln     = (const float*)d_in[12];
  const float* ca_q      = (const float*)d_in[13];
  const float* ca_kv     = (const float*)d_in[14];
  const float* ca_out    = (const float*)d_in[15];
  const float* ca_ff1    = (const float*)d_in[16];
  const float* ca_ff2    = (const float*)d_in[17];
  const float* logit_ln  = (const float*)d_in[18];
  const float* logit_w   = (const float*)d_in[19];
  float* out = (float*)d_out;

  // ---- workspace layout (fp32 elements) ----
  float* ws = (float*)d_ws;
  size_t off = 0;
  auto alloc = [&](size_t n) { float* p = ws + off; off += (n + 31) & ~(size_t)31; return p; };
  float* aA    = alloc((size_t)kM * kD);
  float* aB    = alloc((size_t)kM * kD);
  float* nA    = alloc((size_t)kM * kD);
  float* nB    = alloc((size_t)kM * kD);
  float* xn    = alloc((size_t)kM * kD);
  float* fused = alloc((size_t)kM * kFUSED);     // also reused for ca ff1 output (3088 < 3728)
  float* qr    = alloc((size_t)kM * 512);
  float* kr    = alloc((size_t)kM * 64);
  float* vb    = alloc((size_t)kM * 64);
  float* ao    = alloc((size_t)kM * 512);
  float* ffin  = alloc((size_t)kM * kFF);
  float* kv    = alloc((size_t)kM * 128);

  auto gemm = [&](const float* A, const float* Bw, float* Cc, const float* Rr,
                  int M, int N, int K, int mode) {
    dim3 grid((N + BN - 1) / BN, (M + BM - 1) / BM);
    gemm_wmma_f16<<<grid, 256, 0, stream>>>(A, Bw, Cc, Rr, M, N, K, mode);
  };

  // ---- parallel transformer block: xout = xin + ptb(xin) ----
  auto ptb = [&](const float* xin, float* xout, const float* g,
                 const float* wf, const float* wo, const float* wff) {
    ln_kernel<<<kM, 256, 0, stream>>>(xin, g, xn);
    gemm(xn, wf, fused, nullptr, kM, kFUSED, kD, 0);
    extract_qkv_kernel<<<kM, 64, 0, stream>>>(fused, qr, kr, vb);
    attn_kernel<<<kB * kH * kN, 32, 0, stream>>>(qr, kr, vb, ao, /*causal=*/1, 64, 64);
    silu_mul_kernel<<<((size_t)kM * kFF + 255) / 256, 256, 0, stream>>>(
        fused, ffin, kM, kFF, kFUSED, 640, 640 + kFF);
    gemm(ao, wo, xout, xin, kM, kD, 512, 1);       // xout = xin + attn@Wo
    gemm(ffin, wff, xout, nullptr, kM, kD, kFF, 2);// xout += ff@Wff
  };

  // ---- cross-attn block: xout = xin + cross(xin, ctx) ----
  auto cross = [&](const float* xin, float* xout, const float* ctx, const float* g,
                   const float* wq, const float* wkv, const float* wo,
                   const float* wf1, const float* wf2) {
    ln_kernel<<<kM, 256, 0, stream>>>(xin, g, xn);
    gemm(xn, wq, qr, nullptr, kM, 512, kD, 0);
    gemm(ctx, wkv, kv, nullptr, kM, 128, kD, 0);
    attn_kernel<<<kB * kH * kN, 32, 0, stream>>>(qr, kv, kv + 64, ao, /*causal=*/0, 128, 128);
    gemm(xn, wf1, fused, nullptr, kM, 2 * kFF, kD, 0);
    silu_mul_kernel<<<((size_t)kM * kFF + 255) / 256, 256, 0, stream>>>(
        fused, ffin, kM, kFF, 2 * kFF, 0, kFF);
    gemm(ao, wo, xout, xin, kM, kD, 512, 1);
    gemm(ffin, wf2, xout, nullptr, kM, kD, kFF, 2);
  };

  // ---- pipeline ----
  embed_kernel<<<kB * kS, 256, 0, stream>>>(input, conv_w, conv_b, cls_token, aA, 0);
  embed_kernel<<<kB * kS, 256, 0, stream>>>(input, conv_w, conv_b, cls_token, nA, 1);

  float* acur = aA; float* anext = aB;
  float* ncur = nA; float* nnext = nB;

  for (int i = 0; i < 4; ++i) {
    const float* g   = uni_ln    + (size_t)i * kD;
    const float* wf  = uni_fused + (size_t)i * kD * kFUSED;
    const float* wo  = uni_aout  + (size_t)i * 512 * kD;
    const float* wff = uni_fout  + (size_t)i * kFF * kD;
    ptb(acur, anext, g, wf, wo, wff); { float* t = acur; acur = anext; anext = t; }
    ptb(ncur, nnext, g, wf, wo, wff); { float* t = ncur; ncur = nnext; nnext = t; }
  }
  for (int i = 0; i < 4; ++i) {
    ptb(acur, anext,
        mm_ln    + (size_t)i * kD,
        mm_fused + (size_t)i * kD * kFUSED,
        mm_aout  + (size_t)i * 512 * kD,
        mm_fout  + (size_t)i * kFF * kD);
    { float* t = acur; acur = anext; anext = t; }
    cross(acur, anext, ncur,
          ca_ln  + (size_t)i * kD,
          ca_q   + (size_t)i * kD * 512,
          ca_kv  + (size_t)i * kD * 128,
          ca_out + (size_t)i * 512 * kD,
          ca_ff1 + (size_t)i * kD * 2 * kFF,
          ca_ff2 + (size_t)i * kFF * kD);
    { float* t = acur; acur = anext; anext = t; }
  }

  logit_kernel<<<1, 128, 0, stream>>>(acur, logit_ln, logit_w, out);
}
